// MinCutTAD_GraphClass_19851338842634
// MI455X (gfx1250) — matrix operand order repaired and checked by reading, not compiled, not run on hardware
//
#include <hip/hip_runtime.h>
#include <math.h>

// ---------------------------------------------------------------------------
// MinCutTAD graph classifier for MI455X (gfx1250), wave32 + WMMA f16->f32.
// Heavy GEMMs (S^T A, (S^T A)S, S^T S, adj@x, x@W) use v_wmma_f32_16x16x32_f16
// with f32 accumulation; operands converted f32->f16 while staging to LDS.
// Fragments are loaded from LDS as contiguous b128 chunks (no scalar gathers).
// ---------------------------------------------------------------------------

typedef __attribute__((ext_vector_type(8)))  _Float16 v8h;
typedef __attribute__((ext_vector_type(16))) _Float16 v16h;
typedef __attribute__((ext_vector_type(8)))  float    v8f;
typedef __attribute__((ext_vector_type(4)))  float    v4f;

static constexpr int   N0  = 8192;       // nodes
static constexpr int   EDG = N0 * 32;    // edges
static constexpr int   FIN = 10;         // input features
static constexpr int   HD  = 64;         // hidden
static constexpr int   K0  = 4096;       // pool 0 clusters
static constexpr int   K1  = 2048;       // pool 1 clusters
static constexpr float EPSV = 1e-15f;

// scalar accumulator slots in workspace
enum { S_NUM0 = 0, S_DEN0, S_FRO0, S_ORT0, S_NUM1, S_DEN1, S_FRO1, S_ORT1, S_COUNT = 16 };

// ---------------------------------------------------------------------------
// WMMA GEMM: C[M,N] = op(A) @ B   (row-major C, ldc = N)
//   op(A)(m,k) = TA ? A[k*lda + m] : A[m*lda + k]
//   B(k,n)     = B[k*ldb + n]
// Block: 256 threads (8 waves), tile 128(M) x 64(N), BK = 32.
// Each wave computes a 16x64 strip => 4 wmma per K-step.
// NOTE: every launch in this network has M%128==0, N%64==0, K%32==0, so the
// inner loops carry no bounds checks.
// ---------------------------------------------------------------------------
#define BM 128
#define BN 64
#define BK 32
#define APAD 8   // row stride (BK+APAD)*2B = 80B, multiple of 16B -> b128-aligned

template <bool TA>
__global__ __launch_bounds__(256)
void gemm_wmma(const float* __restrict__ A, const float* __restrict__ B,
               float* __restrict__ C, int M, int N, int K, int lda, int ldb)
{
    // A tile: row-major; a lane's fragment = two contiguous 16B runs of a row.
    __shared__ alignas(16) _Float16 As[BM][BK + APAD];
    // B tile staged in fragment order: [subtile(4)][lane(32)][half(16)]
    __shared__ alignas(16) _Float16 BsF[4 * 32 * 16];

    const int tid  = threadIdx.x;
    const int wave = tid >> 5;
    const int lane = tid & 31;
    const int rowBase = blockIdx.y * BM;
    const int colBase = blockIdx.x * BN;

    // B staging role for this thread: fills 8 halves of one fragment slot
    const int bslot = tid >> 1;                 // 0..127 = subtile*32 + lane
    const int bhg   = tid & 1;                  // half-group (0: K 8h..8h+7, 1: K 16+8h..16+8h+7; h = lane half)
    const int bn    = colBase + ((bslot >> 5) << 4) + (bslot & 15);
    const int bkb   = (bhg << 4) + (((bslot & 31) >> 4) << 3);  // 0 / 8 / 16 / 24

    v8f acc[4] = {};   // 4 sub-tiles of 16 columns

    for (int k0 = 0; k0 < K; k0 += BK) {
        // prefetch next K-tile while we stage this one (global_prefetch_b8)
        if (k0 + BK < K && lane == 0) {
            __builtin_prefetch(&B[(size_t)(k0 + BK) * ldb + colBase], 0, 1);
            if (TA) __builtin_prefetch(&A[(size_t)(k0 + BK) * lda + rowBase], 0, 1);
            else    __builtin_prefetch(&A[(size_t)rowBase * lda + (k0 + BK)], 0, 1);
        }
        __syncthreads();

        // ---- stage A tile (BM x BK) ----
        if (!TA) {
            // vectorized: v4f global load, 4-half LDS store; coalesced over K
            for (int i = tid; i < (BM * BK) / 4; i += 256) {
                const int r  = i >> 3;            // BK/4 = 8 vec4 per row
                const int c4 = (i & 7) << 2;
                const v4f v = *(const v4f*)&A[(size_t)(rowBase + r) * lda + (k0 + c4)];
                As[r][c4 + 0] = (_Float16)v[0];
                As[r][c4 + 1] = (_Float16)v[1];
                As[r][c4 + 2] = (_Float16)v[2];
                As[r][c4 + 3] = (_Float16)v[3];
            }
        } else {
            // transposed read: iterate column-major so lanes coalesce over rows
            for (int i = tid; i < BM * BK; i += 256) {
                const int c = i >> 7;             // i / BM
                const int r = i & (BM - 1);       // i % BM
                As[r][c] = (_Float16)A[(size_t)(k0 + c) * lda + (rowBase + r)];
            }
        }

        // ---- stage B tile (BK x BN) directly in fragment order ----
        {
            v8h pk;
            #pragma unroll
            for (int i = 0; i < 8; ++i)
                pk[i] = (_Float16)B[(size_t)(k0 + bkb + i) * ldb + bn];
            *(v8h*)&BsF[bslot * 16 + bhg * 8] = pk;   // one ds_store_b128
        }
        __syncthreads();

        // ---- A fragment: 2x ds_load_b128 from a row-major row ----
        const int mrow  = wave * 16 + (lane & 15);
        const int kbase = (lane >> 4) * 8;        // lanes 0-15: K0-7/16-23, 16-31: K8-15/24-31
        union { v16h v; v8h h[2]; } au;
        au.h[0] = *(const v8h*)&As[mrow][kbase];
        au.h[1] = *(const v8h*)&As[mrow][kbase + 16];

        #pragma unroll
        for (int t = 0; t < 4; ++t) {
            // ---- B fragment: contiguous 32B chunk (2x ds_load_b128) ----
            const v16h bf = *(const v16h*)&BsF[(t * 32 + lane) * 16];
            acc[t] = __builtin_amdgcn_wmma_f32_16x16x32_f16(
                false, au.v, false, bf, (short)0, acc[t], false, false);
        }
    }

    // C layout: lane 0-15 -> M = vgpr (0..7), lane 16-31 -> M = 8+vgpr; N = lane%16
    const int mrow0 = rowBase + wave * 16 + ((lane >> 4) << 3);
    #pragma unroll
    for (int t = 0; t < 4; ++t) {
        const int gcol = colBase + t * 16 + (lane & 15);
        #pragma unroll
        for (int v = 0; v < 8; ++v)
            C[(size_t)(mrow0 + v) * N + gcol] = acc[t][v];
    }
}

// ---------------------------------------------------------------------------
// Sparse graph build: agg[dst] += X[src]; A[src,dst] += w
// ---------------------------------------------------------------------------
__global__ void scatter_agg_kernel(const float* __restrict__ X,
                                   const int* __restrict__ src,
                                   const int* __restrict__ dst,
                                   float* __restrict__ agg, int ne)
{
    int e = blockIdx.x * blockDim.x + threadIdx.x;
    if (e >= ne) return;
    const int s = src[e], d = dst[e];
    #pragma unroll
    for (int f = 0; f < FIN; ++f)
        atomicAdd(&agg[(size_t)d * FIN + f], X[(size_t)s * FIN + f]);
}

__global__ void scatter_adj_kernel(const int* __restrict__ src,
                                   const int* __restrict__ dst,
                                   const float* __restrict__ w,
                                   float* __restrict__ A, int ne)
{
    int e = blockIdx.x * blockDim.x + threadIdx.x;
    if (e >= ne) return;
    atomicAdd(&A[(size_t)src[e] * N0 + dst[e]], w[e]);
}

// first conv: x = relu(agg @ w_rel0 + b_rel0 + X @ w_root0)   (K = 10, scalar)
__global__ void conv0_kernel(const float* __restrict__ agg, const float* __restrict__ X,
                             const float* __restrict__ w_rel, const float* __restrict__ b_rel,
                             const float* __restrict__ w_root, float* __restrict__ out)
{
    int i = blockIdx.x * blockDim.x + threadIdx.x;
    if (i >= N0 * HD) return;
    const int n = i / HD, h = i % HD;
    float acc = b_rel[h];
    #pragma unroll
    for (int f = 0; f < FIN; ++f)
        acc += agg[(size_t)n * FIN + f] * w_rel[f * HD + h]
             + X[(size_t)n * FIN + f]  * w_root[f * HD + h];
    out[i] = fmaxf(acc, 0.0f);
}

// ---------------------------------------------------------------------------
// softmax over rows (fused bias), also produces per-row sum of squares
// ---------------------------------------------------------------------------
__global__ void softmax_bias_rows(float* __restrict__ p, const float* __restrict__ bias,
                                  float* __restrict__ sumsq, int k)
{
    __shared__ float red[256];
    const int row = blockIdx.x, tid = threadIdx.x;
    float* r = p + (size_t)row * k;

    float m = -3.4e38f;
    for (int c = tid; c < k; c += 256) m = fmaxf(m, r[c] + bias[c]);
    red[tid] = m; __syncthreads();
    for (int s = 128; s > 0; s >>= 1) {
        if (tid < s) red[tid] = fmaxf(red[tid], red[tid + s]);
        __syncthreads();
    }
    m = red[0]; __syncthreads();

    float sum = 0.0f;
    for (int c = tid; c < k; c += 256) { float e = __expf(r[c] + bias[c] - m); r[c] = e; sum += e; }
    red[tid] = sum; __syncthreads();
    for (int s = 128; s > 0; s >>= 1) { if (tid < s) red[tid] += red[tid + s]; __syncthreads(); }
    const float inv = 1.0f / red[0]; __syncthreads();

    float sq = 0.0f;
    for (int c = tid; c < k; c += 256) { float v = r[c] * inv; r[c] = v; sq += v * v; }
    red[tid] = sq; __syncthreads();
    for (int s = 128; s > 0; s >>= 1) { if (tid < s) red[tid] += red[tid + s]; __syncthreads(); }
    if (tid == 0) sumsq[row] = red[0];
}

__global__ void row_sum_kernel(const float* __restrict__ M_, float* __restrict__ out, int ncols)
{
    __shared__ float red[256];
    const int row = blockIdx.x, tid = threadIdx.x;
    const float* r = M_ + (size_t)row * ncols;
    float s = 0.0f;
    for (int c = tid; c < ncols; c += 256) s += r[c];
    red[tid] = s; __syncthreads();
    for (int k = 128; k > 0; k >>= 1) { if (tid < k) red[tid] += red[tid + k]; __syncthreads(); }
    if (tid == 0) out[row] = red[0];
}

// trace(M) -> atomic scalar
__global__ void trace_kernel(const float* __restrict__ M_, int k, float* __restrict__ out)
{
    __shared__ float red[256];
    const int tid = threadIdx.x;
    float s = 0.0f;
    for (int i = blockIdx.x * 256 + tid; i < k; i += gridDim.x * 256)
        s += M_[(size_t)i * k + i];
    red[tid] = s; __syncthreads();
    for (int j = 128; j > 0; j >>= 1) { if (tid < j) red[tid] += red[tid + j]; __syncthreads(); }
    if (tid == 0) atomicAdd(out, red[0]);
}

// dot(a, b) -> atomic scalar
__global__ void dot_kernel(const float* __restrict__ a, const float* __restrict__ b,
                           int n, float* __restrict__ out)
{
    __shared__ float red[256];
    const int tid = threadIdx.x;
    float s = 0.0f;
    for (int i = blockIdx.x * 256 + tid; i < n; i += gridDim.x * 256) s += a[i] * b[i];
    red[tid] = s; __syncthreads();
    for (int j = 128; j > 0; j >>= 1) { if (tid < j) red[tid] += red[tid + j]; __syncthreads(); }
    if (tid == 0) atomicAdd(out, red[0]);
}

// sum of squares -> atomic scalar (Frobenius^2)
__global__ void sumsq_kernel(const float* __restrict__ a, size_t n, float* __restrict__ out)
{
    __shared__ float red[256];
    const int tid = threadIdx.x;
    float s = 0.0f;
    for (size_t i = (size_t)blockIdx.x * 256 + tid; i < n; i += (size_t)gridDim.x * 256)
        s += a[i] * a[i];
    red[tid] = s; __syncthreads();
    for (int j = 128; j > 0; j >>= 1) { if (tid < j) red[tid] += red[tid + j]; __syncthreads(); }
    if (tid == 0) atomicAdd(out, red[0]);
}

// sum_{ij} (SS_ij / ||SS||_F - I_ij / sqrt(k))^2 -> atomic scalar
__global__ void ortho_kernel(const float* __restrict__ ss, const float* __restrict__ froSq,
                             int k, float* __restrict__ out)
{
    __shared__ float red[256];
    const int tid = threadIdx.x;
    const float invn = 1.0f / sqrtf(*froSq);
    const float dk = 1.0f / sqrtf((float)k);
    const size_t tot = (size_t)k * k;
    float s = 0.0f;
    for (size_t i = (size_t)blockIdx.x * 256 + tid; i < tot; i += (size_t)gridDim.x * 256) {
        const int r = (int)(i / k), c = (int)(i % k);
        const float d = ss[i] * invn - ((r == c) ? dk : 0.0f);
        s += d * d;
    }
    red[tid] = s; __syncthreads();
    for (int j = 128; j > 0; j >>= 1) { if (tid < j) red[tid] += red[tid + j]; __syncthreads(); }
    if (tid == 0) atomicAdd(out, red[0]);
}

// zero diagonal, then dsq[row] = sqrt(rowsum) + eps
__global__ void zerodiag_rowsum_kernel(float* __restrict__ A, float* __restrict__ dsq, int k)
{
    __shared__ float red[256];
    const int row = blockIdx.x, tid = threadIdx.x;
    float* r = A + (size_t)row * k;
    if (tid == 0) r[row] = 0.0f;
    __syncthreads();
    float s = 0.0f;
    for (int c = tid; c < k; c += 256) s += r[c];
    red[tid] = s; __syncthreads();
    for (int j = 128; j > 0; j >>= 1) { if (tid < j) red[tid] += red[tid + j]; __syncthreads(); }
    if (tid == 0) dsq[row] = sqrtf(red[0]) + EPSV;
}

// A[i,j] /= dsq[i]*dsq[j]
__global__ void norm_adj_kernel(float* __restrict__ A, const float* __restrict__ dsq, int k)
{
    const size_t tot = (size_t)k * k;
    for (size_t i = (size_t)blockIdx.x * blockDim.x + threadIdx.x; i < tot;
         i += (size_t)gridDim.x * blockDim.x) {
        const int r = (int)(i / k), c = (int)(i % k);
        A[i] = A[i] / (dsq[r] * dsq[c]);
    }
}

// out = (a + b + bias[col])  with optional relu; ncols = HD
__global__ void add3_bias_kernel(const float* __restrict__ a, const float* __restrict__ b,
                                 const float* __restrict__ bias, float* __restrict__ out,
                                 int n, int relu)
{
    int i = blockIdx.x * blockDim.x + threadIdx.x;
    if (i >= n) return;
    float v = a[i] + b[i] + bias[i % HD];
    out[i] = relu ? fmaxf(v, 0.0f) : v;
}

// column mean over M rows of [M, HD]
__global__ void colmean_kernel(const float* __restrict__ x, float* __restrict__ out, int m)
{
    __shared__ float red[256];
    const int col = blockIdx.x, tid = threadIdx.x;
    float s = 0.0f;
    for (int r = tid; r < m; r += 256) s += x[(size_t)r * HD + col];
    red[tid] = s; __syncthreads();
    for (int j = 128; j > 0; j >>= 1) { if (tid < j) red[tid] += red[tid + j]; __syncthreads(); }
    if (tid == 0) out[col] = red[0] / (float)m;
}

// MLP head + loss assembly: d_out = [logit0, logit1, mincut_loss, ortho_loss]
__global__ void head_kernel(const float* __restrict__ xm,
                            const float* __restrict__ w1, const float* __restrict__ b1,
                            const float* __restrict__ w2, const float* __restrict__ b2,
                            const float* __restrict__ scal, float* __restrict__ out)
{
    if (threadIdx.x != 0 || blockIdx.x != 0) return;
    float h[HD];
    for (int j = 0; j < HD; ++j) {
        float t = b1[j];
        for (int i = 0; i < HD; ++i) t += xm[i] * w1[i * HD + j];
        h[j] = fmaxf(t, 0.0f);
    }
    for (int c = 0; c < 2; ++c) {
        float t = b2[c];
        for (int j = 0; j < HD; ++j) t += h[j] * w2[j * 2 + c];
        out[c] = t;
    }
    out[2] = -(scal[S_NUM0] / scal[S_DEN0]) - (scal[S_NUM1] / scal[S_DEN1]);
    out[3] = sqrtf(scal[S_ORT0]) + sqrtf(scal[S_ORT1]);
}

// ---------------------------------------------------------------------------
// host driver
// ---------------------------------------------------------------------------
static inline void launch_gemm(bool ta, const float* A, const float* B, float* C,
                               int M, int N, int K, int lda, int ldb, hipStream_t st)
{
    dim3 grid((N + BN - 1) / BN, (M + BM - 1) / BM);
    if (ta) gemm_wmma<true ><<<grid, 256, 0, st>>>(A, B, C, M, N, K, lda, ldb);
    else    gemm_wmma<false><<<grid, 256, 0, st>>>(A, B, C, M, N, K, lda, ldb);
}

extern "C" void kernel_launch(void* const* d_in, const int* in_sizes, int n_in,
                              void* d_out, int out_size, void* d_ws, size_t ws_size,
                              hipStream_t stream)
{
    (void)in_sizes; (void)n_in; (void)out_size; (void)ws_size;

    const float* X       = (const float*)d_in[0];
    const int*   ei      = (const int*)  d_in[1];
    const float* eattr   = (const float*)d_in[2];
    const float* w_rel0  = (const float*)d_in[3];
    const float* b_rel0  = (const float*)d_in[4];
    const float* w_root0 = (const float*)d_in[5];
    const float* w_rel1  = (const float*)d_in[6];
    const float* b_rel1  = (const float*)d_in[7];
    const float* w_root1 = (const float*)d_in[8];
    const float* w_rel2  = (const float*)d_in[9];
    const float* b_rel2  = (const float*)d_in[10];
    const float* w_root2 = (const float*)d_in[11];
    const float* w_p0    = (const float*)d_in[12];
    const float* b_p0    = (const float*)d_in[13];
    const float* w_p1    = (const float*)d_in[14];
    const float* b_p1    = (const float*)d_in[15];
    const float* w_l1    = (const float*)d_in[16];
    const float* b_l1    = (const float*)d_in[17];
    const float* w_l2    = (const float*)d_in[18];
    const float* b_l2    = (const float*)d_in[19];
    const int* src = ei;            // edge_index[0]
    const int* dst = ei + EDG;      // edge_index[1]
    float* out = (float*)d_out;

    // --- workspace arena (floats) ---
    float* w = (float*)d_ws;
    size_t off = 0;
    auto carve = [&](size_t n) { float* p = w + off; off += n; return p; };

    float* scal = carve(S_COUNT);
    float* agg  = carve((size_t)N0 * FIN);
    float* Xh0  = carve((size_t)N0 * HD);
    float* A0   = carve((size_t)N0 * N0);          // dense adjacency (256 MB)
    float* S0   = carve((size_t)N0 * K0);          // logits -> softmax (in place)
    float* STA0 = carve((size_t)K0 * N0);          // S0^T A ; later reused for SS0
    float* OA0  = carve((size_t)K0 * K0);          // S0^T A S0 -> normalized adj1 (in place)
    float* out0 = carve((size_t)K0 * HD);
    float* d0   = carve(N0);
    float* sq0  = carve(N0);
    float* ds0  = carve(K0);
    float* t1   = carve((size_t)K0 * HD);
    float* t2   = carve((size_t)K0 * HD);
    float* t3   = carve((size_t)K0 * HD);
    float* x1   = carve((size_t)K0 * HD);
    float* S1   = carve((size_t)K0 * K1);
    float* STA1 = carve((size_t)K1 * K0);          // later reused for SS1
    float* OA1  = carve((size_t)K1 * K1);
    float* out1 = carve((size_t)K1 * HD);
    float* d1   = carve(K0);
    float* sq1  = carve(K0);
    float* ds1  = carve(K1);
    float* t4   = carve((size_t)K1 * HD);
    float* t5   = carve((size_t)K1 * HD);
    float* t6   = carve((size_t)K1 * HD);
    float* x2   = carve((size_t)K1 * HD);
    float* xm   = carve(HD);
    float* SS0  = STA0;   // alias: STA0 dead once OA0 is computed
    float* SS1  = STA1;

    // --- zero what must be zero every call (graph-capture safe) ---
    hipMemsetAsync(scal, 0, S_COUNT * sizeof(float), stream);
    hipMemsetAsync(agg,  0, (size_t)N0 * FIN * sizeof(float), stream);
    hipMemsetAsync(A0,   0, (size_t)N0 * N0 * sizeof(float), stream);

    // --- sparse graph build + first conv ---
    scatter_agg_kernel<<<(EDG + 255) / 256, 256, 0, stream>>>(X, src, dst, agg, EDG);
    scatter_adj_kernel<<<(EDG + 255) / 256, 256, 0, stream>>>(src, dst, eattr, A0, EDG);
    conv0_kernel<<<(N0 * HD + 255) / 256, 256, 0, stream>>>(agg, X, w_rel0, b_rel0, w_root0, Xh0);

    // --- level 0 pool ---
    launch_gemm(false, Xh0, w_p0, S0, N0, K0, HD, HD, K0, stream);          // logits
    softmax_bias_rows<<<N0, 256, 0, stream>>>(S0, b_p0, sq0, K0);           // S0, ||s||^2
    row_sum_kernel<<<N0, 256, 0, stream>>>(A0, d0, N0);                     // degrees
    launch_gemm(true,  S0, Xh0, out0, K0, HD, N0, K0, HD, stream);          // S0^T X
    launch_gemm(true,  S0, A0,  STA0, K0, N0, N0, K0, N0, stream);          // S0^T A
    launch_gemm(false, STA0, S0, OA0, K0, K0, N0, N0, K0, stream);          // (S0^T A) S0
    trace_kernel<<<16, 256, 0, stream>>>(OA0, K0, &scal[S_NUM0]);
    dot_kernel<<<32, 256, 0, stream>>>(d0, sq0, N0, &scal[S_DEN0]);
    launch_gemm(true,  S0, S0, SS0, K0, K0, N0, K0, K0, stream);            // S0^T S0
    sumsq_kernel<<<2048, 256, 0, stream>>>(SS0, (size_t)K0 * K0, &scal[S_FRO0]);
    ortho_kernel<<<2048, 256, 0, stream>>>(SS0, &scal[S_FRO0], K0, &scal[S_ORT0]);
    zerodiag_rowsum_kernel<<<K0, 256, 0, stream>>>(OA0, ds0, K0);
    norm_adj_kernel<<<2048, 256, 0, stream>>>(OA0, ds0, K0);                // adj1 in place

    // --- level 1 conv + pool ---
    launch_gemm(false, OA0, out0, t1, K0, HD, K0, K0, HD, stream);          // adj1 @ x
    launch_gemm(false, t1, w_rel1, t2, K0, HD, HD, HD, HD, stream);
    launch_gemm(false, out0, w_root1, t3, K0, HD, HD, HD, HD, stream);
    add3_bias_kernel<<<(K0 * HD + 255) / 256, 256, 0, stream>>>(t2, t3, b_rel1, x1, K0 * HD, 1);

    launch_gemm(false, x1, w_p1, S1, K0, K1, HD, HD, K1, stream);
    softmax_bias_rows<<<K0, 256, 0, stream>>>(S1, b_p1, sq1, K1);
    row_sum_kernel<<<K0, 256, 0, stream>>>(OA0, d1, K0);
    launch_gemm(true,  S1, x1,  out1, K1, HD, K0, K1, HD, stream);
    launch_gemm(true,  S1, OA0, STA1, K1, K0, K0, K1, K0, stream);
    launch_gemm(false, STA1, S1, OA1, K1, K1, K0, K0, K1, stream);
    trace_kernel<<<8, 256, 0, stream>>>(OA1, K1, &scal[S_NUM1]);
    dot_kernel<<<16, 256, 0, stream>>>(d1, sq1, K0, &scal[S_DEN1]);
    launch_gemm(true,  S1, S1, SS1, K1, K1, K0, K1, K1, stream);
    sumsq_kernel<<<1024, 256, 0, stream>>>(SS1, (size_t)K1 * K1, &scal[S_FRO1]);
    ortho_kernel<<<1024, 256, 0, stream>>>(SS1, &scal[S_FRO1], K1, &scal[S_ORT1]);
    zerodiag_rowsum_kernel<<<K1, 256, 0, stream>>>(OA1, ds1, K1);
    norm_adj_kernel<<<1024, 256, 0, stream>>>(OA1, ds1, K1);                // adj2 in place

    // --- final conv (no relu) + readout ---
    launch_gemm(false, OA1, out1, t4, K1, HD, K1, K1, HD, stream);
    launch_gemm(false, t4, w_rel2, t5, K1, HD, HD, HD, HD, stream);
    launch_gemm(false, out1, w_root2, t6, K1, HD, HD, HD, HD, stream);
    add3_bias_kernel<<<(K1 * HD + 255) / 256, 256, 0, stream>>>(t5, t6, b_rel2, x2, K1 * HD, 0);
    colmean_kernel<<<HD, 256, 0, stream>>>(x2, xm, K1);
    head_kernel<<<1, 1, 0, stream>>>(xm, w_l1, b_l1, w_l2, b_l2, scal, out);
}
